// Decoder_17978733101275
// MI455X (gfx1250) — matrix operand order repaired and checked by reading, not compiled, not run on hardware
//
#include <hip/hip_runtime.h>
#include <hip/hip_bf16.h>
#include <math.h>

// ---------------------------------------------------------------------------
// Problem constants (V,E,H,B,S from the reference)
// ---------------------------------------------------------------------------
#define VSZ 32000
#define ESZ 512
#define HSZ 1024
#define BSZ 64
#define SSZ 1024

typedef __attribute__((ext_vector_type(16))) __bf16 v16bf;
typedef __attribute__((ext_vector_type(8)))  __bf16 v8bf;
typedef __attribute__((ext_vector_type(8)))  float  v8f;
typedef __attribute__((ext_vector_type(4)))  int    v4i;

#define AS1 __attribute__((address_space(1)))
#define AS3 __attribute__((address_space(3)))

// ---------------------------------------------------------------------------
// WMMA fragment loaders (16x16x32 bf16, wave32; this file's contiguous k order, same for A and B)
// A (16x32, 16-bit): lane L -> row L%16 ; half = L/16 ;
//   k = half*16 + 0..15
// B (32x16, 16-bit): lane L -> col L%16 ; half = L/16 ; k = half*16 + 0..15
// C/D (16x16 f32)  : VGPR i, lane L -> m = i + 8*(L/16), n = L%16
// ---------------------------------------------------------------------------
__device__ inline v16bf load_a_frag_bf16(const __bf16* __restrict__ row,
                                         int k0, int half) {
    v8bf lo = *(const v8bf*)(row + k0 + half * 16);
    v8bf hi = *(const v8bf*)(row + k0 + half * 16 + 8);
    return __builtin_shufflevector(lo, hi, 0, 1, 2, 3, 4, 5, 6, 7,
                                   8, 9, 10, 11, 12, 13, 14, 15);
}

// 16 consecutive f32 -> bf16 fragment chunk (B fragment for one lane)
__device__ inline v16bf load_b_frag_f32(const float* __restrict__ p) {
    v16bf r;
#pragma unroll
    for (int i = 0; i < 16; ++i) r[i] = (__bf16)p[i];
    return r;
}

__device__ inline v8f wmma_bf16(v16bf a, v16bf b, v8f c) {
    return __builtin_amdgcn_wmma_f32_16x16x32_bf16(
        /*neg_a=*/false, a, /*neg_b=*/false, b,
        /*c_mod=*/(short)0, c, /*reuse_a=*/false, /*reuse_b=*/false);
}

// fast transcendental helpers (v_exp_f32 + v_rcp_f32)
__device__ inline float fast_tanh(float x) {
    x = fminf(15.0f, fmaxf(-15.0f, x));       // avoid inf-inf
    float e = __expf(2.0f * x);
    return (e - 1.0f) * __builtin_amdgcn_rcpf(e + 1.0f);
}
__device__ inline float fast_sigmoid(float x) {
    return __builtin_amdgcn_rcpf(1.0f + __expf(-x));
}

// ---------------------------------------------------------------------------
// K0: embedding gather -> x[:, 0:E].  grid (E/256, B), block 256
// ---------------------------------------------------------------------------
__global__ __launch_bounds__(256) void embed_kernel(
    const int* __restrict__ input, const float* __restrict__ emb,
    float* __restrict__ xbuf) {
    int b = blockIdx.y;
    int e = blockIdx.x * 256 + threadIdx.x;
    xbuf[(size_t)b * (ESZ + HSZ) + e] = emb[(size_t)input[b] * ESZ + e];
}

// ---------------------------------------------------------------------------
// K1: bias[b,h] = attn_b[h] + h_last[b,:] . attn_W[h, 0:H]
// block = 4 h-rows x 64 batches. grid H/4 = 256. float4 loads.
// ---------------------------------------------------------------------------
__global__ __launch_bounds__(256) void attn_bias_kernel(
    const float* __restrict__ hidden, const float* __restrict__ attn_W,
    const float* __restrict__ attn_b, float* __restrict__ bias) {
    int t = threadIdx.x;
    int b = t & 63;
    int h = blockIdx.x * 4 + (t >> 6);
    const float4* wr = (const float4*)(attn_W + (size_t)h * (2 * HSZ));
    const float4* hl = (const float4*)(hidden + (size_t)b * HSZ);
    float acc = attn_b[h];
#pragma unroll 4
    for (int k = 0; k < HSZ / 4; ++k) {
        float4 a = hl[k], w = wr[k];
        acc = fmaf(a.x, w.x, acc); acc = fmaf(a.y, w.y, acc);
        acc = fmaf(a.z, w.z, acc); acc = fmaf(a.w, w.w, acc);
    }
    bias[(size_t)b * HSZ + h] = acc;
}

// ---------------------------------------------------------------------------
// K1b: one-shot bf16 conversion of attn_W[:, H:2H] -> w2bf [H][H]
// (2 MB; removes per-WMMA f32->bf16 cvt + halves B-side L2 bytes in K2)
// grid H, block 256
// ---------------------------------------------------------------------------
__global__ __launch_bounds__(256) void w2cvt_kernel(
    const float* __restrict__ attn_W, __bf16* __restrict__ w2bf) {
    int n = blockIdx.x;
    int t = threadIdx.x;
    float4 f = ((const float4*)(attn_W + (size_t)n * (2 * HSZ) + HSZ))[t];
    __bf16 tmp[4] = {(__bf16)f.x, (__bf16)f.y, (__bf16)f.z, (__bf16)f.w};
    *(uint2*)(w2bf + (size_t)n * HSZ + t * 4) = *(const uint2*)tmp;
}

// ---------------------------------------------------------------------------
// K2: fused attention-energy GEMM + tanh + dot(v)  ->  scores[B,S]
// scores[b,s] = sum_h v[h] * tanh( bias[b,h] + sum_k enc[b,s,k]*W2[h,k] )
// One block = 64 rows (one batch). 8 waves = 4 M-subtiles x 2 N-halves.
// enc tile staged to LDS as bf16 once; W2 read as bf16 from L2.
// ---------------------------------------------------------------------------
#define SK_PAD 8
__global__ __launch_bounds__(256) void attn_scores_kernel(
    const float* __restrict__ enc, const __bf16* __restrict__ w2bf,
    const float* __restrict__ bias, const float* __restrict__ vvec,
    float* __restrict__ scores) {
    __shared__ __bf16 Abuf[64][HSZ + SK_PAD];   // 132 KB (<= 320 KB/WGP)
    __shared__ float  sred[2][64];

    int tile = blockIdx.x;               // 1024 tiles of 64 rows
    int b = tile >> 4;                   // S/64 = 16 tiles per batch
    int s0 = (tile & 15) * 64;
    int t = threadIdx.x;

    // Stage A: 64 rows x 1024 f32 -> bf16 LDS (packed ds_store_b64)
    const float* encb = enc + ((size_t)b * SSZ + s0) * HSZ;
#pragma unroll
    for (int it = 0; it < 64; ++it) {
        int i = it * 256 + t;            // float4 index, 64*256 total
        int row = i >> 8;
        int c4 = i & 255;
        float4 f = ((const float4*)(encb + (size_t)row * HSZ))[c4];
        __bf16 tmp[4] = {(__bf16)f.x, (__bf16)f.y, (__bf16)f.z, (__bf16)f.w};
        *(uint2*)(&Abuf[row][c4 * 4]) = *(const uint2*)tmp;
    }
    __syncthreads();

    int wave = t >> 5, lane = t & 31;
    int sub = wave & 3;                  // which 16-row M subtile
    int nh = wave >> 2;                  // n-half (0/1)
    int lrow = lane & 15, lhalf = lane >> 4;
    const __bf16* arow = &Abuf[sub * 16 + lrow][0];

    float sc[8];
#pragma unroll
    for (int i = 0; i < 8; ++i) sc[i] = 0.0f;

    for (int j = 0; j < 32; ++j) {       // 32 N-chunks of 16 per wave
        int n0 = nh * 512 + j * 16;
        int n = n0 + lrow;               // this lane's output column
        const __bf16* wrow = w2bf + (size_t)n * HSZ;
        v8f acc = {0.f, 0.f, 0.f, 0.f, 0.f, 0.f, 0.f, 0.f};
#pragma unroll 4
        for (int k0 = 0; k0 < HSZ; k0 += 32) {
            v16bf a = load_a_frag_bf16(arow, k0, lhalf);
            v16bf bf = *(const v16bf*)(wrow + k0 + lhalf * 16);
            acc = wmma_bf16(a, bf, acc);
        }
        float bn = bias[(size_t)b * HSZ + n];
        float vn = vvec[n];
#pragma unroll
        for (int i = 0; i < 8; ++i) sc[i] += vn * fast_tanh(acc[i] + bn);
    }

    // reduce over the 16 lanes of each half (columns) -> per-row sums
#pragma unroll
    for (int i = 0; i < 8; ++i) {
        float x = sc[i];
        x += __shfl_xor(x, 8, 32);
        x += __shfl_xor(x, 4, 32);
        x += __shfl_xor(x, 2, 32);
        x += __shfl_xor(x, 1, 32);
        sc[i] = x;
    }
    if (lrow == 0) {
#pragma unroll
        for (int i = 0; i < 8; ++i)
            sred[nh][sub * 16 + lhalf * 8 + i] = sc[i];
    }
    __syncthreads();
    if (t < 64)
        scores[(size_t)b * SSZ + s0 + t] = sred[0][t] + sred[1][t];
}

// ---------------------------------------------------------------------------
// K3: softmax over S per batch. grid B, block 256 (4 elems/thread)
// ---------------------------------------------------------------------------
__global__ __launch_bounds__(256) void softmax_kernel(
    const float* __restrict__ scores, float* __restrict__ wts) {
    __shared__ float red[8];
    __shared__ float stat;
    int b = blockIdx.x, t = threadIdx.x;
    int wave = t >> 5, lane = t & 31;
    const float* s = scores + (size_t)b * SSZ;
    float4 x = ((const float4*)s)[t];

    float m = fmaxf(fmaxf(x.x, x.y), fmaxf(x.z, x.w));
    for (int msk = 16; msk >= 1; msk >>= 1) m = fmaxf(m, __shfl_xor(m, msk, 32));
    if (lane == 0) red[wave] = m;
    __syncthreads();
    if (t == 0) {
        float mm = red[0];
#pragma unroll
        for (int i = 1; i < 8; ++i) mm = fmaxf(mm, red[i]);
        stat = mm;
    }
    __syncthreads();
    float bm = stat;

    float4 e;
    e.x = __expf(x.x - bm); e.y = __expf(x.y - bm);
    e.z = __expf(x.z - bm); e.w = __expf(x.w - bm);
    float sum = e.x + e.y + e.z + e.w;
    for (int msk = 16; msk >= 1; msk >>= 1) sum += __shfl_xor(sum, msk, 32);
    __syncthreads();
    if (lane == 0) red[wave] = sum;
    __syncthreads();
    if (t == 0) {
        float ss = 0.f;
#pragma unroll
        for (int i = 0; i < 8; ++i) ss += red[i];
        stat = 1.0f / ss;
    }
    __syncthreads();
    float inv = stat;
    float4 o = {e.x * inv, e.y * inv, e.z * inv, e.w * inv};
    ((float4*)(wts + (size_t)b * SSZ))[t] = o;
}

// ---------------------------------------------------------------------------
// K4: context[b,h] = sum_s w[b,s]*enc[b,s,h]  (fully coalesced over h)
// Weight row (4 KB) staged to LDS with CDNA5 async global->LDS copy.
// grid (H/256, B), block 256
// ---------------------------------------------------------------------------
__global__ __launch_bounds__(256) void context_kernel(
    const float* __restrict__ wts, const float* __restrict__ enc,
    float* __restrict__ xbuf, __bf16* __restrict__ ybf) {
    __shared__ float wl[SSZ];
    int b = blockIdx.y;
    int t = threadIdx.x;

    const float* wsrc = wts + (size_t)b * SSZ + t * 4;   // 16B per thread
#if __has_builtin(__builtin_amdgcn_global_load_async_to_lds_b128)
    __builtin_amdgcn_global_load_async_to_lds_b128(
        (AS1 v4i*)wsrc, (AS3 v4i*)(&wl[t * 4]), 0, 0);
#if __has_builtin(__builtin_amdgcn_s_wait_asynccnt)
    __builtin_amdgcn_s_wait_asynccnt(0);
#else
    asm volatile("s_wait_asynccnt 0x0" ::: "memory");
#endif
#else
    *(float4*)(&wl[t * 4]) = *(const float4*)wsrc;
#endif
    __syncthreads();

    int h = blockIdx.x * 256 + t;
    const float* e = enc + (size_t)b * SSZ * HSZ + h;
    float acc = 0.f;
#pragma unroll 4
    for (int s = 0; s < SSZ; s += 4) {
        float4 w4 = *(const float4*)(&wl[s]);
        acc = fmaf(w4.x, e[(size_t)(s + 0) * HSZ], acc);
        acc = fmaf(w4.y, e[(size_t)(s + 1) * HSZ], acc);
        acc = fmaf(w4.z, e[(size_t)(s + 2) * HSZ], acc);
        acc = fmaf(w4.w, e[(size_t)(s + 3) * HSZ], acc);
    }
    xbuf[(size_t)b * (ESZ + HSZ) + ESZ + h] = acc;
    ybf[(size_t)b * (2 * HSZ) + HSZ + h] = (__bf16)acc;
}

// ---------------------------------------------------------------------------
// K5: single-step LSTM. block = 4 h x 64 b (weight rows broadcast over b).
// grid H/4 = 256. float4 weight/activation loads.
// ---------------------------------------------------------------------------
__global__ __launch_bounds__(256) void lstm_kernel(
    const float* __restrict__ xbuf, const float* __restrict__ hidden,
    const float* __restrict__ cell, const float* __restrict__ W_ih,
    const float* __restrict__ W_hh, const float* __restrict__ b_ih,
    const float* __restrict__ b_hh, float* __restrict__ out_h,
    float* __restrict__ out_c, __bf16* __restrict__ ybf) {
    int t = threadIdx.x;
    int b = t & 63;
    int h = blockIdx.x * 4 + (t >> 6);
    const int KX = ESZ + HSZ;            // 1536

    float gi = b_ih[h]           + b_hh[h];
    float gf = b_ih[HSZ + h]     + b_hh[HSZ + h];
    float gg = b_ih[2 * HSZ + h] + b_hh[2 * HSZ + h];
    float go = b_ih[3 * HSZ + h] + b_hh[3 * HSZ + h];

    {
        const float4* x  = (const float4*)(xbuf + (size_t)b * KX);
        const float4* ri = (const float4*)(W_ih + (size_t)h * KX);
        const float4* rf = (const float4*)(W_ih + (size_t)(HSZ + h) * KX);
        const float4* rg = (const float4*)(W_ih + (size_t)(2 * HSZ + h) * KX);
        const float4* ro = (const float4*)(W_ih + (size_t)(3 * HSZ + h) * KX);
#pragma unroll 2
        for (int k = 0; k < KX / 4; ++k) {
            float4 xv = x[k];
            float4 a = ri[k], c = rf[k], d = rg[k], e = ro[k];
            gi = fmaf(xv.x, a.x, gi); gi = fmaf(xv.y, a.y, gi);
            gi = fmaf(xv.z, a.z, gi); gi = fmaf(xv.w, a.w, gi);
            gf = fmaf(xv.x, c.x, gf); gf = fmaf(xv.y, c.y, gf);
            gf = fmaf(xv.z, c.z, gf); gf = fmaf(xv.w, c.w, gf);
            gg = fmaf(xv.x, d.x, gg); gg = fmaf(xv.y, d.y, gg);
            gg = fmaf(xv.z, d.z, gg); gg = fmaf(xv.w, d.w, gg);
            go = fmaf(xv.x, e.x, go); go = fmaf(xv.y, e.y, go);
            go = fmaf(xv.z, e.z, go); go = fmaf(xv.w, e.w, go);
        }
    }
    {
        const float4* hl = (const float4*)(hidden + (size_t)b * HSZ);
        const float4* qi = (const float4*)(W_hh + (size_t)h * HSZ);
        const float4* qf = (const float4*)(W_hh + (size_t)(HSZ + h) * HSZ);
        const float4* qg = (const float4*)(W_hh + (size_t)(2 * HSZ + h) * HSZ);
        const float4* qo = (const float4*)(W_hh + (size_t)(3 * HSZ + h) * HSZ);
#pragma unroll 2
        for (int k = 0; k < HSZ / 4; ++k) {
            float4 hv = hl[k];
            float4 a = qi[k], c = qf[k], d = qg[k], e = qo[k];
            gi = fmaf(hv.x, a.x, gi); gi = fmaf(hv.y, a.y, gi);
            gi = fmaf(hv.z, a.z, gi); gi = fmaf(hv.w, a.w, gi);
            gf = fmaf(hv.x, c.x, gf); gf = fmaf(hv.y, c.y, gf);
            gf = fmaf(hv.z, c.z, gf); gf = fmaf(hv.w, c.w, gf);
            gg = fmaf(hv.x, d.x, gg); gg = fmaf(hv.y, d.y, gg);
            gg = fmaf(hv.z, d.z, gg); gg = fmaf(hv.w, d.w, gg);
            go = fmaf(hv.x, e.x, go); go = fmaf(hv.y, e.y, go);
            go = fmaf(hv.z, e.z, go); go = fmaf(hv.w, e.w, go);
        }
    }
    float iv = fast_sigmoid(gi), fv = fast_sigmoid(gf), ov = fast_sigmoid(go);
    float gv = fast_tanh(gg);
    float c = fv * cell[(size_t)b * HSZ + h] + iv * gv;
    float hn = ov * fast_tanh(c);
    out_h[(size_t)b * HSZ + h] = hn;
    out_c[(size_t)b * HSZ + h] = c;
    ybf[(size_t)b * (2 * HSZ) + h] = (__bf16)hn;
}

// ---------------------------------------------------------------------------
// K6: FC projection  pred[B, V] = y[B, 2H] @ W_fc^T + b_fc   (WMMA bf16)
// M=64 (4 subtiles per wave), 16 cols per wave, 128 cols per block,
// grid = V/128 = 250. W_fc streamed ONCE from HBM (f32 -> bf16 in-register)
// -> pure 262 MB HBM roofline (~11 us at 23.3 TB/s).
// ---------------------------------------------------------------------------
__global__ __launch_bounds__(256) void fc_kernel(
    const __bf16* __restrict__ ybf, const float* __restrict__ W_fc,
    const float* __restrict__ b_fc, float* __restrict__ pred) {
    int t = threadIdx.x;
    int wave = t >> 5, lane = t & 31;
    int lrow = lane & 15, lhalf = lane >> 4;
    int n0 = blockIdx.x * 128 + wave * 16;
    int n = n0 + lrow;
    const float* wrow = W_fc + (size_t)n * (2 * HSZ);

    v8f acc[4];
#pragma unroll
    for (int s = 0; s < 4; ++s)
        acc[s] = (v8f){0.f, 0.f, 0.f, 0.f, 0.f, 0.f, 0.f, 0.f};

#pragma unroll 2
    for (int k0 = 0; k0 < 2 * HSZ; k0 += 32) {
        v16bf bf = load_b_frag_f32(wrow + k0 + lhalf * 16);
#pragma unroll
        for (int s = 0; s < 4; ++s) {
            const __bf16* arow = ybf + (size_t)(s * 16 + lrow) * (2 * HSZ);
            v16bf a = load_a_frag_bf16(arow, k0, lhalf);
            acc[s] = wmma_bf16(a, bf, acc[s]);
        }
    }
    float bn = b_fc[n];
#pragma unroll
    for (int s = 0; s < 4; ++s) {
#pragma unroll
        for (int i = 0; i < 8; ++i) {
            int m = s * 16 + lhalf * 8 + i;
            pred[(size_t)m * VSZ + n] = acc[s][i] + bn;
        }
    }
}

// ---------------------------------------------------------------------------
// Host launcher
// ---------------------------------------------------------------------------
extern "C" void kernel_launch(void* const* d_in, const int* in_sizes, int n_in,
                              void* d_out, int out_size, void* d_ws,
                              size_t ws_size, hipStream_t stream) {
    (void)in_sizes; (void)n_in; (void)out_size; (void)ws_size;
    const int*   input  = (const int*)d_in[0];
    const float* hidden = (const float*)d_in[1];   // [1,B,H] -> [B,H]
    const float* cell   = (const float*)d_in[2];
    const float* enc    = (const float*)d_in[3];   // [B,S,H]
    const float* emb    = (const float*)d_in[4];
    const float* attn_W = (const float*)d_in[5];   // [H, 2H]
    const float* attn_b = (const float*)d_in[6];
    const float* vvec   = (const float*)d_in[7];
    const float* W_ih   = (const float*)d_in[8];
    const float* W_hh   = (const float*)d_in[9];
    const float* b_ih   = (const float*)d_in[10];
    const float* b_hh   = (const float*)d_in[11];
    const float* W_fc   = (const float*)d_in[12];
    const float* b_fc   = (const float*)d_in[13];

    // workspace layout (bytes)
    char* ws = (char*)d_ws;
    float*  bias   = (float*)(ws + 0);                         // B*H f32
    float*  scores = (float*)(ws + 262144);                    // B*S f32
    float*  wts    = (float*)(ws + 524288);                    // B*S f32
    float*  xbuf   = (float*)(ws + 786432);                    // B*(E+H) f32
    __bf16* ybf    = (__bf16*)(ws + 1179648);                  // B*2H bf16
    __bf16* w2bf   = (__bf16*)(ws + 1441792);                  // H*H bf16 (2MB)

    float* pred  = (float*)d_out;                              // [B, V]
    float* out_h = pred + (size_t)BSZ * VSZ;                   // [1,B,H]
    float* out_c = out_h + (size_t)BSZ * HSZ;                  // [1,B,H]

    embed_kernel<<<dim3(ESZ / 256, BSZ), 256, 0, stream>>>(input, emb, xbuf);
    attn_bias_kernel<<<HSZ / 4, 256, 0, stream>>>(hidden, attn_W, attn_b, bias);
    w2cvt_kernel<<<HSZ, 256, 0, stream>>>(attn_W, w2bf);
    attn_scores_kernel<<<(BSZ * SSZ) / 64, 256, 0, stream>>>(enc, w2bf, bias,
                                                             vvec, scores);
    softmax_kernel<<<BSZ, 256, 0, stream>>>(scores, wts);
    context_kernel<<<dim3(HSZ / 256, BSZ), 256, 0, stream>>>(wts, enc, xbuf,
                                                             ybf);
    lstm_kernel<<<HSZ / 4, 256, 0, stream>>>(xbuf, hidden, cell, W_ih, W_hh,
                                             b_ih, b_hh, out_h, out_c, ybf);
    fc_kernel<<<VSZ / 128, 256, 0, stream>>>(ybf, W_fc, b_fc, pred);
}
